// StressGCN_Conv_28724741275672
// MI455X (gfx1250) — matrix operand order repaired and checked by reading, not compiled, not run on hardware
//
#include <hip/hip_runtime.h>

#define NNODES 100000
#define NEDGES 1600000
#define INC 128
#define HIDC 256

typedef __attribute__((ext_vector_type(16))) _Float16 v16h;
typedef __attribute__((ext_vector_type(4)))  _Float16 h4;
typedef __attribute__((ext_vector_type(8)))  float    v8f;
typedef __attribute__((ext_vector_type(4)))  float    f4;
typedef int i4v __attribute__((vector_size(4 * sizeof(int))));

#define AS1 __attribute__((address_space(1)))
#define AS3 __attribute__((address_space(3)))

union AFrag {
  v16h h;
  struct { f4 lo, hi; } f;
};

// ---------------- small utility kernels ----------------

__global__ void k_fill(float* __restrict__ p, float v, int n) {
  int i = blockIdx.x * blockDim.x + threadIdx.x;
  if (i < n) p[i] = v;
}

__global__ void k_deg(const int* __restrict__ col, float* __restrict__ deg, int e) {
  int i = blockIdx.x * blockDim.x + threadIdx.x;
  if (i < e) atomicAdd(&deg[col[i]], 1.0f);
}

__global__ void k_rsqrt(const float* __restrict__ deg, float* __restrict__ dis, int n) {
  int i = blockIdx.x * blockDim.x + threadIdx.x;
  if (i < n) dis[i] = rsqrtf(deg[i]);   // self-loops guarantee deg >= 1
}

__global__ void k_cvt(const float* __restrict__ x, _Float16* __restrict__ y, int n) {
  int i = blockIdx.x * blockDim.x + threadIdx.x;
  if (i < n) y[i] = (_Float16)x[i];
}

// Pre-convert a [K x NOUT] f32 weight into f16, tiled as [kt][n][kk]
// (kt = K/32 tiles, n = output col, kk = k within tile). One-time cost.
__global__ void k_cvtw(const float* __restrict__ W, _Float16* __restrict__ wt,
                       int K, int NOUT) {
  int o = blockIdx.x * blockDim.x + threadIdx.x;
  if (o >= K * NOUT) return;
  int tile = NOUT * 32;
  int kt = o / tile;
  int r  = o % tile;
  int n  = r >> 5;
  int kk = r & 31;
  int k  = kt * 32 + kk;
  wt[o] = (_Float16)W[(size_t)k * NOUT + n];
}

// 32 lanes per edge; lane handles channels [lane*4, lane*4+4) and +128.
__global__ void k_scatter(const int* __restrict__ row, const int* __restrict__ col,
                          const float* __restrict__ hws, float* __restrict__ agg, int e) {
  int t = blockIdx.x * blockDim.x + threadIdx.x;
  int ei = t >> 5;
  int lane = t & 31;
  if (ei >= e) return;
  size_t bs = (size_t)row[ei] * HIDC;
  size_t bd = (size_t)col[ei] * HIDC;
  int c0 = lane * 4;
  f4 v0 = *(const f4*)(hws + bs + c0);
  f4 v1 = *(const f4*)(hws + bs + c0 + 128);
  atomicAdd(&agg[bd + c0 + 0], v0.x);
  atomicAdd(&agg[bd + c0 + 1], v0.y);
  atomicAdd(&agg[bd + c0 + 2], v0.z);
  atomicAdd(&agg[bd + c0 + 3], v0.w);
  atomicAdd(&agg[bd + c0 + 128], v1.x);
  atomicAdd(&agg[bd + c0 + 129], v1.y);
  atomicAdd(&agg[bd + c0 + 130], v1.z);
  atomicAdd(&agg[bd + c0 + 131], v1.w);
}

// ---------------- WMMA GEMM ----------------
// A: [M x K] row-major f16. WT: pre-tiled f16 weights [K/32][NOUT][32].
// out = act((A @ W) * rowscale + bias); optional duplicate f32 store (agg seed).
template <int K, int NOUT, bool RELU, bool SCALE, bool BIAS, bool OUTF32, bool DUP>
__global__ __launch_bounds__(256) void gemm_wmma(
    const _Float16* __restrict__ A, const _Float16* __restrict__ WT,
    const float* __restrict__ bias, const float* __restrict__ rowscale,
    float* __restrict__ outF, float* __restrict__ outF2,
    _Float16* __restrict__ outH, int M) {
  constexpr int NT = NOUT / 16;
  constexpr int CH = NOUT * 4;          // 16-byte chunks per 32-K weight tile
  __shared__ _Float16 ldsW[NOUT * 32];  // [n][kk]

  const int tid   = threadIdx.x;
  const int wave  = tid >> 5;
  const int lane  = tid & 31;
  const int half_ = lane >> 4;
  const int l16   = lane & 15;
  const int mBase = blockIdx.x * 128 + wave * 16;
  const int arow_i = mBase + l16;
  const int arow_c = arow_i < M ? arow_i : (M - 1);
  const _Float16* arow = A + (size_t)arow_c * K;

  v8f acc[NT];
#pragma unroll
  for (int t = 0; t < NT; ++t)
#pragma unroll
    for (int r = 0; r < 8; ++r) acc[t][r] = 0.0f;

  for (int k0 = 0; k0 < K; k0 += 32) {
    const _Float16* wsrc = WT + (size_t)(k0 >> 5) * (NOUT * 32);
    __syncthreads();  // protect previous iteration's LDS reads
#if __has_builtin(__builtin_amdgcn_global_load_async_to_lds_b128)
#pragma unroll
    for (int i = 0; i < CH / 256; ++i) {
      int c = tid + i * 256;
      __builtin_amdgcn_global_load_async_to_lds_b128(
          (AS1 i4v*)(wsrc + c * 8), (AS3 i4v*)(ldsW + c * 8), 0, 0);
    }
#if __has_builtin(__builtin_amdgcn_s_wait_asynccnt)
    __builtin_amdgcn_s_wait_asynccnt(0);
#else
    asm volatile("s_wait_asynccnt 0x0" ::: "memory");
#endif
#else
#pragma unroll
    for (int i = 0; i < CH / 256; ++i) {
      int c = tid + i * 256;
      *(f4*)(ldsW + c * 8) = *(const f4*)(wsrc + c * 8);
    }
#endif
    __syncthreads();

    // A fragment (16-bit A 16x32 layout: lane half 0 -> K {0..7,16..23}, half 1 -> +8).
    AFrag a;
    a.f.lo = *(const f4*)(arow + k0 + 8 * half_);
    a.f.hi = *(const f4*)(arow + k0 + 16 + 8 * half_);

#pragma unroll
    for (int t = 0; t < NT; ++t) {
      // B fragment (32x16): lane half 0 -> K {0..7,16..23}, half 1 -> +8, N = lane%16.
      AFrag b;
      const _Float16* bp = &ldsW[(t * 16 + l16) * 32 + 8 * half_];
      b.f.lo = *(const f4*)(bp);
      b.f.hi = *(const f4*)(bp + 16);
      acc[t] = __builtin_amdgcn_wmma_f32_16x16x32_f16(
          false, a.h, false, b.h, (short)0, acc[t], false, false);
    }
  }

  // Epilogue. C/D layout: VGPR r -> row mBase + r + 8*half, col = t*16 + lane%16.
  const int rbase = mBase + 8 * half_;
  float rs[8];
#pragma unroll
  for (int r = 0; r < 8; ++r) {
    int rr = rbase + r;
    rs[r] = SCALE ? rowscale[rr < M ? rr : 0] : 1.0f;
  }
#pragma unroll
  for (int t = 0; t < NT; ++t) {
    int c = t * 16 + l16;
    float bv = BIAS ? bias[c] : 0.0f;
#pragma unroll
    for (int r = 0; r < 8; ++r) {
      int rr = rbase + r;
      if (rr < M) {
        float v = acc[t][r];
        if (SCALE) v *= rs[r];
        if (BIAS)  v += bv;
        if (RELU)  v = fmaxf(v, 0.0f);
        if (OUTF32) {
          outF[(size_t)rr * NOUT + c] = v;
          if (DUP) outF2[(size_t)rr * NOUT + c] = v;
        } else {
          outH[(size_t)rr * NOUT + c] = (_Float16)v;
        }
      }
    }
  }
}

// ---------------- per-node: h = relu(LN(agg*dis + conv_b)) -> f16 ----------------
__global__ __launch_bounds__(256) void k_post(
    const float* __restrict__ agg, const float* __restrict__ dis,
    const float* __restrict__ cb, const float* __restrict__ g,
    const float* __restrict__ bb, _Float16* __restrict__ act, int n_nodes) {
  int wave = threadIdx.x >> 5, lane = threadIdx.x & 31;
  int n = blockIdx.x * 8 + wave;
  if (n >= n_nodes) return;
  float d = dis[n];
  size_t base = (size_t)n * HIDC;
  int c0 = lane * 4;  // channels c0..c0+3 and c0+128..c0+131
  f4 a0 = *(const f4*)(agg + base + c0);
  f4 a1 = *(const f4*)(agg + base + c0 + 128);
  f4 cb0 = *(const f4*)(cb + c0);
  f4 cb1 = *(const f4*)(cb + c0 + 128);
  float v[8];
  v[0] = a0.x * d + cb0.x; v[1] = a0.y * d + cb0.y;
  v[2] = a0.z * d + cb0.z; v[3] = a0.w * d + cb0.w;
  v[4] = a1.x * d + cb1.x; v[5] = a1.y * d + cb1.y;
  v[6] = a1.z * d + cb1.z; v[7] = a1.w * d + cb1.w;
  float s = 0.0f, s2 = 0.0f;
#pragma unroll
  for (int k = 0; k < 8; ++k) { s += v[k]; s2 += v[k] * v[k]; }
#pragma unroll
  for (int off = 16; off > 0; off >>= 1) {
    s  += __shfl_xor(s, off, 32);
    s2 += __shfl_xor(s2, off, 32);
  }
  float mu  = s * (1.0f / HIDC);
  float var = s2 * (1.0f / HIDC) - mu * mu;
  float inv = rsqrtf(var + 1e-5f);
  f4 g0 = *(const f4*)(g + c0);
  f4 g1 = *(const f4*)(g + c0 + 128);
  f4 b0 = *(const f4*)(bb + c0);
  f4 b1 = *(const f4*)(bb + c0 + 128);
  h4 o0, o1;
  o0.x = (_Float16)fmaxf((v[0] - mu) * inv * g0.x + b0.x, 0.0f);
  o0.y = (_Float16)fmaxf((v[1] - mu) * inv * g0.y + b0.y, 0.0f);
  o0.z = (_Float16)fmaxf((v[2] - mu) * inv * g0.z + b0.z, 0.0f);
  o0.w = (_Float16)fmaxf((v[3] - mu) * inv * g0.w + b0.w, 0.0f);
  o1.x = (_Float16)fmaxf((v[4] - mu) * inv * g1.x + b1.x, 0.0f);
  o1.y = (_Float16)fmaxf((v[5] - mu) * inv * g1.y + b1.y, 0.0f);
  o1.z = (_Float16)fmaxf((v[6] - mu) * inv * g1.z + b1.z, 0.0f);
  o1.w = (_Float16)fmaxf((v[7] - mu) * inv * g1.w + b1.w, 0.0f);
  *(h4*)(act + base + c0) = o0;
  *(h4*)(act + base + c0 + 128) = o1;
}

// ---------------- head: out[n] = dot(t[n,0:128], lin2_w) + lin2_b ----------------
__global__ __launch_bounds__(256) void k_head(
    const float* __restrict__ tmat, const float* __restrict__ w2,
    const float* __restrict__ b2, float* __restrict__ out, int n_nodes) {
  int wave = threadIdx.x >> 5, lane = threadIdx.x & 31;
  int n = blockIdx.x * 8 + wave;
  if (n >= n_nodes) return;
  size_t base = (size_t)n * 128;
  int c0 = lane * 4;
  f4 tv = *(const f4*)(tmat + base + c0);
  f4 wv = *(const f4*)(w2 + c0);
  float s = tv.x * wv.x + tv.y * wv.y + tv.z * wv.z + tv.w * wv.w;
#pragma unroll
  for (int off = 16; off > 0; off >>= 1) s += __shfl_xor(s, off, 32);
  if (lane == 0) out[n] = s + b2[0];
}

// ---------------- host orchestration ----------------
extern "C" void kernel_launch(void* const* d_in, const int* in_sizes, int n_in,
                              void* d_out, int out_size, void* d_ws, size_t ws_size,
                              hipStream_t stream) {
  (void)in_sizes; (void)n_in; (void)out_size; (void)ws_size;
  const int N = NNODES, E = NEDGES;
  const float* x      = (const float*)d_in[0];
  const int*   ei     = (const int*)d_in[1];   // [2][E]
  const float* enc_w  = (const float*)d_in[2];
  const float* enc_b  = (const float*)d_in[3];
  const float* conv_w = (const float*)d_in[4]; // [4][256][256]
  const float* conv_b = (const float*)d_in[5]; // [4][256]
  const float* ln_g   = (const float*)d_in[6];
  const float* ln_b   = (const float*)d_in[7];
  const float* lin1_w = (const float*)d_in[8];
  const float* lin1_b = (const float*)d_in[9];
  const float* lin2_w = (const float*)d_in[10];
  const float* lin2_b = (const float*)d_in[11];
  float* out = (float*)d_out;
  const int* row = ei;
  const int* col = ei + E;

  char* ws = (char*)d_ws;
  size_t off = 0;
  auto carve = [&](size_t bytes) -> void* {
    void* p = ws + off;
    off = (off + bytes + 255) & ~(size_t)255;
    return p;
  };
  float*    dis  = (float*)carve((size_t)N * 4);
  _Float16* x16  = (_Float16*)carve((size_t)N * INC * 2);
  _Float16* act  = (_Float16*)carve((size_t)N * HIDC * 2);
  float*    hws  = (float*)carve((size_t)N * HIDC * 4);
  float*    agg  = (float*)carve((size_t)N * HIDC * 4);
  _Float16* wtE  = (_Float16*)carve((size_t)INC * HIDC * 2);        // encoder
  _Float16* wtC  = (_Float16*)carve((size_t)4 * HIDC * HIDC * 2);   // conv x4
  _Float16* wtL  = (_Float16*)carve((size_t)HIDC * 128 * 2);        // lin1
  float*    deg  = hws;  // reuse hws before any GEMM writes it

  dim3 b256(256);
  // degree + symmetric norm (self-loop gives deg >= 1)
  k_fill<<<(N + 255) / 256, b256, 0, stream>>>(deg, 1.0f, N);
  k_deg<<<(E + 255) / 256, b256, 0, stream>>>(col, deg, E);
  k_rsqrt<<<(N + 255) / 256, b256, 0, stream>>>(deg, dis, N);
  k_cvt<<<(N * INC + 255) / 256, b256, 0, stream>>>(x, x16, N * INC);

  // one-time weight conversion to pre-tiled f16
  k_cvtw<<<(INC * HIDC + 255) / 256, b256, 0, stream>>>(enc_w, wtE, INC, HIDC);
  for (int i = 0; i < 4; ++i)
    k_cvtw<<<(HIDC * HIDC + 255) / 256, b256, 0, stream>>>(
        conv_w + (size_t)i * HIDC * HIDC, wtC + (size_t)i * HIDC * HIDC, HIDC, HIDC);
  k_cvtw<<<(HIDC * 128 + 255) / 256, b256, 0, stream>>>(lin1_w, wtL, HIDC, 128);

  const int gblocks = (N + 127) / 128;
  // encoder: h0 = x @ enc_w + enc_b  -> f16 activations
  gemm_wmma<INC, HIDC, false, false, true, false, false><<<gblocks, b256, 0, stream>>>(
      x16, wtE, enc_b, nullptr, nullptr, nullptr, act, N);

  for (int i = 0; i < 4; ++i) {
    // hws = agg = (h @ conv_w[i]) * dis[row]   (agg seeded with self-loop term)
    gemm_wmma<HIDC, HIDC, false, true, false, true, true><<<gblocks, b256, 0, stream>>>(
        act, wtC + (size_t)i * HIDC * HIDC, nullptr, dis, hws, agg, nullptr, N);
    k_scatter<<<(E * 32 + 255) / 256, b256, 0, stream>>>(row, col, hws, agg, E);
    k_post<<<(N + 7) / 8, b256, 0, stream>>>(
        agg, dis, conv_b + i * HIDC, ln_g + i * HIDC, ln_b + i * HIDC, act, N);
  }

  // t = relu(h @ lin1_w + lin1_b) -> f32 in hws
  gemm_wmma<HIDC, 128, true, false, true, true, false><<<gblocks, b256, 0, stream>>>(
      act, wtL, lin1_b, nullptr, hws, nullptr, nullptr, N);
  // out = t @ lin2_w + lin2_b
  k_head<<<(N + 7) / 8, b256, 0, stream>>>(hws, lin2_w, lin2_b, out, N);
}
